// ConvSelfAttention_18141941858864
// MI455X (gfx1250) — matrix operand order repaired and checked by reading, not compiled, not run on hardware
//
#include <hip/hip_runtime.h>
#include <hip/hip_bf16.h>

typedef __attribute__((ext_vector_type(16))) __bf16 v16bf;
typedef __attribute__((ext_vector_type(8)))  float  v8f;

#define BB 8
#define CC 64
#define NN 4096          // H*W = 64*64

union V16 {
    v16bf v;
    uint4 q[2];
    unsigned u[8];
};

// Pack two f32 -> packed bf16 pair with one v_perm_b32 (+2 rounding adds).
__device__ inline unsigned pack_bf16(float lo, float hi) {
    unsigned a = __float_as_uint(lo) + 0x8000u;   // round-to-nearest
    unsigned b = __float_as_uint(hi) + 0x8000u;
    return __builtin_amdgcn_perm(b, a, 0x07060302u); // bytes [b3 b2 a3 a2]
}
__device__ inline unsigned short f2bf(float x) {
    return (unsigned short)((__float_as_uint(x) + 0x8000u) >> 16);
}

// Build bf16 A operand (16x32) from 16 contiguous f32 rows of a 64-wide matrix.
// A k order used here (contiguous, B packed the same way): lanes0-15 hold K{kb..kb+15} with kb=chunk*32,
// lanes16-31 hold K{kb+16..kb+31}. Row = lane%16 for all lanes.
__device__ inline void load_a_rows_f32(const float* xr, int hi, V16 a[2]) {
#pragma unroll
    for (int kc = 0; kc < 2; ++kc) {
        int kb = kc * 32 + hi * 16;
        float4 r0 = *(const float4*)(xr + kb);
        float4 r1 = *(const float4*)(xr + kb + 4);
        float4 r2 = *(const float4*)(xr + kb + 8);
        float4 r3 = *(const float4*)(xr + kb + 12);
        a[kc].u[0] = pack_bf16(r0.x, r0.y);
        a[kc].u[1] = pack_bf16(r0.z, r0.w);
        a[kc].u[2] = pack_bf16(r1.x, r1.y);
        a[kc].u[3] = pack_bf16(r1.z, r1.w);
        a[kc].u[4] = pack_bf16(r2.x, r2.y);
        a[kc].u[5] = pack_bf16(r2.z, r2.w);
        a[kc].u[6] = pack_bf16(r3.x, r3.y);
        a[kc].u[7] = pack_bf16(r3.z, r3.w);
    }
}

// ---------------------------------------------------------------------------
// Kernel 1: raw 1x1 convs.  X(32768x64) @ [Kf|Kg|Kh](64x80) via bf16 WMMA.
// One wave = 16 rows x 80 cols. t==0 writes a FUSED fg buffer (f cols 0-7,
// g cols 8-15) so all stores are uniform-addressed.
// ---------------------------------------------------------------------------
__global__ __launch_bounds__(128) void conv_fgh_kernel(
    const float* __restrict__ x,
    const float* __restrict__ kf, const float* __restrict__ kg,
    const float* __restrict__ kh,
    float* __restrict__ fg_raw,   // (B*N, 16)
    float* __restrict__ h_raw)    // (B*N, 64)
{
    const int wave = threadIdx.x >> 5;
    const int lane = threadIdx.x & 31;
    const int lrow = lane & 15;
    const int hi   = lane >> 4;
    const int row0 = (blockIdx.x * 4 + wave) * 16;

    V16 a[2];
    load_a_rows_f32(x + (size_t)(row0 + lrow) * CC, hi, a);

#pragma unroll
    for (int t = 0; t < 5; ++t) {
        const float* W;
        int col, ldw;
        if (t == 0) { W = (lrow < 8) ? kf : kg; col = lrow & 7; ldw = 8; }
        else        { W = kh; col = (t - 1) * 16 + lrow; ldw = 64; }

        V16 bm[2];
#pragma unroll
        for (int kc = 0; kc < 2; ++kc) {
            int kb = kc * 32 + hi * 16;  // B layout: lane half selects K run
#pragma unroll
            for (int e = 0; e < 8; ++e)
                bm[kc].u[e] = pack_bf16(W[(kb + 2 * e) * ldw + col],
                                        W[(kb + 2 * e + 1) * ldw + col]);
        }
        v8f c = {};
        c = __builtin_amdgcn_wmma_f32_16x16x32_bf16(false, a[0].v, false, bm[0].v,
                                                    (short)0, c, false, false);
        c = __builtin_amdgcn_wmma_f32_16x16x32_bf16(false, a[1].v, false, bm[1].v,
                                                    (short)0, c, false, false);
#pragma unroll
        for (int r = 0; r < 8; ++r) {
            int row = row0 + r + 8 * hi;
            if (t == 0) fg_raw[(size_t)row * 16 + lrow] = c[r];
            else        h_raw[(size_t)row * 64 + (t - 1) * 16 + lrow] = c[r];
        }
    }
}

// ---------------------------------------------------------------------------
// Kernel 2a: maxpool(2x2,s1,SAME) on f, pass-through g; pack rows to bf16.
// ---------------------------------------------------------------------------
__global__ __launch_bounds__(256) void pool_fg_kernel(
    const float* __restrict__ fg_raw,
    uint4* __restrict__ fB, uint4* __restrict__ gB)
{
    int idx = blockIdx.x * blockDim.x + threadIdx.x;   // b*N + n
    if (idx >= BB * NN) return;
    int n = idx & (NN - 1);
    int w = n & 63, h = n >> 6;
    bool we = (w < 63), he = (h < 63);

    const float* bf = fg_raw + (size_t)idx * 16;       // f half
    float v[8];
#pragma unroll
    for (int c = 0; c < 8; ++c) {
        float m = bf[c];
        if (we)       m = fmaxf(m, bf[16 + c]);        // n+1
        if (he)       m = fmaxf(m, bf[1024 + c]);      // n+64
        if (we && he) m = fmaxf(m, bf[1040 + c]);      // n+65
        v[c] = m;
    }
    uint4 o;
    o.x = pack_bf16(v[0], v[1]); o.y = pack_bf16(v[2], v[3]);
    o.z = pack_bf16(v[4], v[5]); o.w = pack_bf16(v[6], v[7]);
    fB[idx] = o;

    const float* bg = bf + 8;                          // g half (no pooling)
    o.x = pack_bf16(bg[0], bg[1]); o.y = pack_bf16(bg[2], bg[3]);
    o.z = pack_bf16(bg[4], bg[5]); o.w = pack_bf16(bg[6], bg[7]);
    gB[idx] = o;
}

// ---------------------------------------------------------------------------
// Kernel 2b: maxpool h, PRE-SCALE by 1/sumexp(n), store TRANSPOSED bf16:
// hT[b][c][n] = rowinv[n] * maxpool(h)[n][c].  (Runs AFTER the stats kernel.)
// ---------------------------------------------------------------------------
__global__ __launch_bounds__(256) void pool_h_kernel(
    const float* __restrict__ h_raw, const float* __restrict__ rowinv,
    unsigned short* __restrict__ hT)
{
    int idx = blockIdx.x * blockDim.x + threadIdx.x;   // (b*64 + c)*N + n
    if (idx >= BB * CC * NN) return;
    int n = idx & (NN - 1);
    int c = (idx >> 12) & 63;
    int b = idx >> 18;
    int w = n & 63, h = n >> 6;
    const float* base = h_raw + ((size_t)(b * NN + n)) * 64 + c;
    float m = base[0];
    if (w < 63)           m = fmaxf(m, base[64]);
    if (h < 63)           m = fmaxf(m, base[4096]);
    if (w < 63 && h < 63) m = fmaxf(m, base[4160]);
    hT[idx] = f2bf(m * rowinv[b * NN + n]);
}

// ---------------------------------------------------------------------------
// Kernel 3: per-row softmax stats (max, 1/sumexp) over all 4096 keys.
// Recompute e = f.g (K=8) with LDS-staged g chunks; two sweeps.
// ---------------------------------------------------------------------------
__global__ __launch_bounds__(256) void softmax_stats_kernel(
    const uint4* __restrict__ fB, const uint4* __restrict__ gB,
    float* __restrict__ rowmax, float* __restrict__ rowinv)
{
    __shared__ float gs[256][8];
    const int b = blockIdx.x >> 4;
    const int n = (blockIdx.x & 15) * 256 + threadIdx.x;

    uint4 fq = fB[(size_t)b * NN + n];
    float fr[8];
    fr[0] = __uint_as_float(fq.x << 16); fr[1] = __uint_as_float(fq.x & 0xffff0000u);
    fr[2] = __uint_as_float(fq.y << 16); fr[3] = __uint_as_float(fq.y & 0xffff0000u);
    fr[4] = __uint_as_float(fq.z << 16); fr[5] = __uint_as_float(fq.z & 0xffff0000u);
    fr[6] = __uint_as_float(fq.w << 16); fr[7] = __uint_as_float(fq.w & 0xffff0000u);

    float M = -__builtin_inff();
    for (int mc = 0; mc < 16; ++mc) {
        __syncthreads();
        uint4 gq = gB[(size_t)b * NN + mc * 256 + threadIdx.x];
        gs[threadIdx.x][0] = __uint_as_float(gq.x << 16);
        gs[threadIdx.x][1] = __uint_as_float(gq.x & 0xffff0000u);
        gs[threadIdx.x][2] = __uint_as_float(gq.y << 16);
        gs[threadIdx.x][3] = __uint_as_float(gq.y & 0xffff0000u);
        gs[threadIdx.x][4] = __uint_as_float(gq.z << 16);
        gs[threadIdx.x][5] = __uint_as_float(gq.z & 0xffff0000u);
        gs[threadIdx.x][6] = __uint_as_float(gq.w << 16);
        gs[threadIdx.x][7] = __uint_as_float(gq.w & 0xffff0000u);
        __syncthreads();
#pragma unroll 4
        for (int m = 0; m < 256; ++m) {
            const float* g = gs[m];
            float e = fr[0]*g[0] + fr[1]*g[1] + fr[2]*g[2] + fr[3]*g[3]
                    + fr[4]*g[4] + fr[5]*g[5] + fr[6]*g[6] + fr[7]*g[7];
            M = fmaxf(M, e);
        }
    }
    float S = 0.f;
    for (int mc = 0; mc < 16; ++mc) {
        __syncthreads();
        uint4 gq = gB[(size_t)b * NN + mc * 256 + threadIdx.x];
        gs[threadIdx.x][0] = __uint_as_float(gq.x << 16);
        gs[threadIdx.x][1] = __uint_as_float(gq.x & 0xffff0000u);
        gs[threadIdx.x][2] = __uint_as_float(gq.y << 16);
        gs[threadIdx.x][3] = __uint_as_float(gq.y & 0xffff0000u);
        gs[threadIdx.x][4] = __uint_as_float(gq.z << 16);
        gs[threadIdx.x][5] = __uint_as_float(gq.z & 0xffff0000u);
        gs[threadIdx.x][6] = __uint_as_float(gq.w << 16);
        gs[threadIdx.x][7] = __uint_as_float(gq.w & 0xffff0000u);
        __syncthreads();
#pragma unroll 4
        for (int m = 0; m < 256; ++m) {
            const float* g = gs[m];
            float e = fr[0]*g[0] + fr[1]*g[1] + fr[2]*g[2] + fr[3]*g[3]
                    + fr[4]*g[4] + fr[5]*g[5] + fr[6]*g[6] + fr[7]*g[7];
            S += __expf(e - M);
        }
    }
    rowmax[(size_t)b * NN + n] = M;
    rowinv[(size_t)b * NN + n] = 1.0f / S;
}

// ---------------------------------------------------------------------------
// Kernel 4: main fused pass. Per wave: one 16-wide m-tile, all 64 c.
// Loop n in 32s: e = f@gT (2 WMMA), exp(e-max), per-wave LDS transpose,
// oT[m,c] += attnT @ (h*inv) (4 WMMA). No block barriers: the LDS tile is
// wave-private and DS ops are in-order within a wave. rowmax comes in as
// broadcast float4 loads. attn never touches HBM.
// ---------------------------------------------------------------------------
__global__ __launch_bounds__(128) void attn_kernel(
    const uint4* __restrict__ fB, const uint4* __restrict__ gB,
    const unsigned short* __restrict__ hT,
    const float* __restrict__ rowmax,
    float* __restrict__ o_ws)
{
    __shared__ __align__(16) unsigned attnT[4][256];  // [m(16)][n(32) bf16] per wave

    const int wave = threadIdx.x >> 5;
    const int lane = threadIdx.x & 31;
    const int lrow = lane & 15;
    const int hi   = lane >> 4;

    const int gtile = blockIdx.x * 4 + wave;   // b*256 + m-tile
    const int b  = gtile >> 8;
    const int m0 = (gtile & 255) * 16;

    // Loop-invariant B operand gT (K=8 zero-padded to 32)
    V16 bg;
    {
        uint4 z = make_uint4(0u, 0u, 0u, 0u);
        bg.q[0] = (hi == 0) ? gB[(size_t)b * NN + m0 + lrow] : z;
        bg.q[1] = z;
    }

    v8f acc[4] = {};

    for (int np = 0; np < NN; np += 32) {
        // rowmax for this lane-half's rows: broadcast float4 loads
        const float* mp = rowmax + (size_t)b * NN + np + 8 * hi;
        float4 mA = *(const float4*)(mp);
        float4 mB = *(const float4*)(mp + 4);
        float4 mC = *(const float4*)(mp + 16);
        float4 mD = *(const float4*)(mp + 20);
        float mx1[8] = {mA.x, mA.y, mA.z, mA.w, mB.x, mB.y, mB.z, mB.w};
        float mx2[8] = {mC.x, mC.y, mC.z, mC.w, mD.x, mD.y, mD.z, mD.w};

        // e tiles: rows np..np+15 and np+16..np+31
        V16 a1, a2;
        {
            uint4 z = make_uint4(0u, 0u, 0u, 0u);
            a1.q[0] = (hi == 0) ? fB[(size_t)b * NN + np + lrow] : z;
            a1.q[1] = z;
            a2.q[0] = (hi == 0) ? fB[(size_t)b * NN + np + 16 + lrow] : z;
            a2.q[1] = z;
        }
        v8f ez = {};
        v8f e1 = __builtin_amdgcn_wmma_f32_16x16x32_bf16(false, a1.v, false, bg.v,
                                                         (short)0, ez, false, false);
        v8f e2 = __builtin_amdgcn_wmma_f32_16x16x32_bf16(false, a2.v, false, bg.v,
                                                         (short)0, ez, false, false);

        // exp(e - max) + transposed bf16 spill to LDS: attnT[m][n]
#pragma unroll
        for (int r = 0; r < 8; r += 2) {
            int nl = r + 8 * hi;
            float p0 = __expf(e1[r]     - mx1[r]);
            float p1 = __expf(e1[r + 1] - mx1[r + 1]);
            attnT[wave][lrow * 16 + (nl >> 1)] = pack_bf16(p0, p1);
            float p2 = __expf(e2[r]     - mx2[r]);
            float p3 = __expf(e2[r + 1] - mx2[r + 1]);
            attnT[wave][lrow * 16 + ((nl + 16) >> 1)] = pack_bf16(p2, p3);
        }

        // A operand: attnT (M=m, K=n); contiguous ds_load_b128 pairs per lane
        V16 at;
        {
            int k0 = hi * 8;  // dword offset of this lane-half's first K run
            at.q[0] = *(const uint4*)&attnT[wave][lrow * 16 + k0];
            at.q[1] = *(const uint4*)&attnT[wave][lrow * 16 + k0 + 4];
        }

        // B operand: h*inv (K=n, N=c) from transposed hT — contiguous 32B/lane
#pragma unroll
        for (int ct = 0; ct < 4; ++ct) {
            int c = ct * 16 + lrow;
            const unsigned short* hp =
                hT + (((size_t)(b * CC + c)) << 12) + np + hi * 16;
            V16 bh;
            bh.q[0] = *(const uint4*)(hp);
            bh.q[1] = *(const uint4*)(hp + 8);
            acc[ct] = __builtin_amdgcn_wmma_f32_16x16x32_bf16(
                false, at.v, false, bh.v, (short)0, acc[ct], false, false);
        }
    }

    // D: lane holds c = ct*16+lrow, rows m = m0 + r + 8*hi  ->  o_ws[b][c][m]
#pragma unroll
    for (int ct = 0; ct < 4; ++ct) {
        int c = ct * 16 + lrow;
        float* op = o_ws + (((size_t)(b * CC + c)) << 12) + m0 + 8 * hi;
#pragma unroll
        for (int r = 0; r < 8; ++r) op[r] = acc[ct][r];
    }
}

// ---------------------------------------------------------------------------
// Kernel 5: conv_o + residual.  o_ws flat == x2(32768x64); out = x + g*x2@Ko
// ---------------------------------------------------------------------------
__global__ __launch_bounds__(128) void conv_o_kernel(
    const float* __restrict__ x2, const float* __restrict__ ko,
    const float* __restrict__ xin, const float* __restrict__ gamma,
    float* __restrict__ out)
{
    const int wave = threadIdx.x >> 5;
    const int lane = threadIdx.x & 31;
    const int lrow = lane & 15;
    const int hi   = lane >> 4;
    const int row0 = (blockIdx.x * 4 + wave) * 16;

    V16 a[2];
    load_a_rows_f32(x2 + (size_t)(row0 + lrow) * 64, hi, a);
    float gm = *gamma;

#pragma unroll
    for (int t = 0; t < 4; ++t) {
        int col = t * 16 + lrow;
        V16 bm[2];
#pragma unroll
        for (int kc = 0; kc < 2; ++kc) {
            int kb = kc * 32 + hi * 16;
#pragma unroll
            for (int e = 0; e < 8; ++e)
                bm[kc].u[e] = pack_bf16(ko[(kb + 2 * e) * 64 + col],
                                        ko[(kb + 2 * e + 1) * 64 + col]);
        }
        v8f c = {};
        c = __builtin_amdgcn_wmma_f32_16x16x32_bf16(false, a[0].v, false, bm[0].v,
                                                    (short)0, c, false, false);
        c = __builtin_amdgcn_wmma_f32_16x16x32_bf16(false, a[1].v, false, bm[1].v,
                                                    (short)0, c, false, false);
#pragma unroll
        for (int r = 0; r < 8; ++r) {
            size_t o = (size_t)(row0 + r + 8 * hi) * 64 + col;
            out[o] = xin[o] + gm * c[r];
        }
    }
}

extern "C" void kernel_launch(void* const* d_in, const int* in_sizes, int n_in,
                              void* d_out, int out_size, void* d_ws, size_t ws_size,
                              hipStream_t stream) {
    (void)in_sizes; (void)n_in; (void)out_size; (void)ws_size;
    const float* x  = (const float*)d_in[0];
    const float* kf = (const float*)d_in[1];
    const float* kg = (const float*)d_in[2];
    const float* kh = (const float*)d_in[3];
    const float* ko = (const float*)d_in[4];
    const float* gm = (const float*)d_in[5];
    float* out = (float*)d_out;

    char* ws = (char*)d_ws;
    float*          fg_raw = (float*)(ws + 0);           //  2 MB (B*N,16)
    float*          h_raw  = (float*)(ws + 2097152);     //  8 MB
    uint4*          fB     = (uint4*)(ws + 10485760);    // .5 MB bf16
    uint4*          gB     = (uint4*)(ws + 11010048);    // .5 MB bf16
    unsigned short* hT     = (unsigned short*)(ws + 11534336); // 4 MB bf16 (B,C,N)
    float*          rmax   = (float*)(ws + 15728640);    // 128 KB
    float*          rinv   = (float*)(ws + 15859712);    // 128 KB
    float*          o_ws   = (float*)(ws + 15990784);    //  8 MB (B,C,N)

    conv_fgh_kernel<<<512, 128, 0, stream>>>(x, kf, kg, kh, fg_raw, h_raw);
    pool_fg_kernel<<<128, 256, 0, stream>>>(fg_raw, fB, gB);
    softmax_stats_kernel<<<128, 256, 0, stream>>>(fB, gB, rmax, rinv);
    pool_h_kernel<<<8192, 256, 0, stream>>>(h_raw, rinv, hT);   // uses rinv
    attn_kernel<<<512, 128, 0, stream>>>(fB, gB, hT, rmax, o_ws);
    conv_o_kernel<<<512, 128, 0, stream>>>(o_ws, ko, x, gm, out);
}
